// HiLo_CrossAttention_21371757265020
// MI455X (gfx1250) — matrix operand order repaired and checked by reading, not compiled, not run on hardware
//
#include <hip/hip_runtime.h>

typedef __attribute__((ext_vector_type(16))) _Float16 v16h;
typedef __attribute__((ext_vector_type(8)))  float    v8f;
typedef _Float16 F16;

// ---------------------------------------------------------------------------
// WMMA fragment helpers (CDNA5 wave32 layouts, CDNA5 ISA §7.12.2)
// A (16x32 f16), row-major [M][K]: lane L -> row m = L&15, K base (L>>4)*8;
//   halves 0..7 = K kb..kb+7, halves 8..15 = K kb+16..kb+23 (2 contiguous runs)
// B (32x16 f16) from TRANSPOSED storage Bt[N][K]: lane L -> n = L&15,
//   K base (L>>4)*8; halves 0..7 = K kb..kb+7, halves 8..15 = K kb+16..kb+23 (same as A).
// C/D (16x16 f32): lane L -> col n = L&15, VGPR v -> row m = v + (L>>4)*8
// ---------------------------------------------------------------------------
__device__ __forceinline__ v16h wmma_load_a(const F16* base, int ld) {
  const int lane = threadIdx.x & 31;
  const F16* p = base + (lane & 15) * ld + (lane >> 4) * 8;
  v16h a;
#pragma unroll
  for (int e = 0; e < 8; ++e) {
    a[e]     = p[e];        // K = kb + e
    a[e + 8] = p[16 + e];   // K = kb + 16 + e
  }
  return a;
}

__device__ __forceinline__ v16h wmma_load_bt(const F16* base, int ld) {
  const int lane = threadIdx.x & 31;
  const F16* p = base + (lane & 15) * ld + (lane >> 4) * 8;
  v16h b;
#pragma unroll
  for (int e = 0; e < 8; ++e) { b[e] = p[e]; b[e + 8] = p[16 + e]; }
  return b;
}

#define WMMA_F32_F16(a, b, c) \
  __builtin_amdgcn_wmma_f32_16x16x32_f16(false, (a), false, (b), (short)0, (c), false, false)

// ---------------------------------------------------------------------------
// 2x2 maxpool of both inputs in one pass
// ---------------------------------------------------------------------------
__global__ void maxpool2_kernel(const float* __restrict__ a,
                                const float* __restrict__ b,
                                float* __restrict__ oa,
                                float* __restrict__ ob,
                                int Ho, int Wo, long total) {
  long idx = (long)blockIdx.x * blockDim.x + threadIdx.x;
  if (idx >= total) return;
  int x = (int)(idx % Wo);
  long r = idx / Wo;
  int y = (int)(r % Ho);
  long bc = r / Ho;
  const int Wi = 2 * Wo;
  long base = (bc * (2 * Ho) + 2 * y) * Wi + 2 * x;
  float m0 = fmaxf(a[base], a[base + 1]);
  float m1 = fmaxf(a[base + Wi], a[base + Wi + 1]);
  oa[idx] = fmaxf(m0, m1);
  float n0 = fmaxf(b[base], b[base + 1]);
  float n1 = fmaxf(b[base + Wi], b[base + Wi + 1]);
  ob[idx] = fmaxf(n0, n1);
}

// ---------------------------------------------------------------------------
// One workgroup (8 wave32) per (batch, patch).  All five GEMM stages on WMMA.
// Padded leading dims: LD64=72, LD128=136, weights LDW=72.
// Phased LDS layout (127.5 KB), regions reused as phases retire:
//   [     0,  18432) Pq^T f16 [128c][72]     -> ph2+: V^T f16 [128c][72]
//   [ 18432,  36864) Pk^T f16 [128c][72]     -> ph3+: Qh f16 [64j][136]
//   [ 36864,  64512) W q/k/v f16 3x[64][72]  -> ph3+: Kh f16 [64j][136] @36864
//                                               ph5+: attn f16 [64][72] @54272
//   [ 64512,  97280) Qraw f32 [64][128]      -> ph4+: S f32 [64][64]
//   [ 97280, 130048) Kraw f32 [64][128]
//   [130048, 130560) invNormQ[64], invNormK[64] f32
// ---------------------------------------------------------------------------
#define LD64  72
#define LD128 136

__launch_bounds__(256)
__global__ void attn_branch_kernel(const float* __restrict__ xq,   // q source (x_down)
                                   const float* __restrict__ xk,   // k/v source (x_up)
                                   const float* __restrict__ Wq,
                                   const float* __restrict__ Wk,
                                   const float* __restrict__ Wv,
                                   const float* __restrict__ temp,
                                   float* __restrict__ out,
                                   int H, int Hp) {
  __shared__ __align__(16) unsigned char smem[130560];
  F16*   sPQt  = (F16*)(smem);                // [128c][LD64], valid k=i<64
  F16*   sPKt  = (F16*)(smem + 18432);
  F16*   sW    = (F16*)(smem + 36864);        // 3 x [64][LD64]
  float* sQF   = (float*)(smem + 64512);      // [64][128]
  float* sKF   = (float*)(smem + 97280);
  F16*   sVt   = (F16*)(smem);                // ph2+: [128c][LD64], k=j
  F16*   sQh   = (F16*)(smem + 18432);        // ph3+: [64j][LD128]
  F16*   sKh   = (F16*)(smem + 36864);        // ph3+: [64j][LD128]
  F16*   sAttn = (F16*)(smem + 54272);        // ph5+: [64][LD64]
  float* sS    = (float*)(smem + 64512);      // ph4+: [64][64]
  float* sInvQ = (float*)(smem + 130048);
  float* sInvK = (float*)(smem + 130304);

  const int hw   = Hp * Hp;
  const int b    = blockIdx.x / hw;
  const int nidx = blockIdx.x % hw;
  const int h    = nidx / Hp;
  const int w    = nidx % Hp;
  const int t    = threadIdx.x;
  const int wave = t >> 5;
  const int lane = t & 31;
  const float tempv = temp[0];

  // ---- phase 0: stage dilated patches (f32->f16, transposed) + weights ---
  for (int idx = t; idx < 64 * 128; idx += 256) {
    int i = idx & 63, c = idx >> 6;          // i fast -> contiguous LDS rows
    int pi = i >> 3, pj = i & 7;
    long g = ((long)(b * 128 + c) * H + (pi * Hp + h)) * H + (pj * Hp + w);
    sPQt[c * LD64 + i] = (F16)xq[g];
    sPKt[c * LD64 + i] = (F16)xk[g];
  }
  for (int idx = t; idx < 3 * 4096; idx += 256) {
    const float* src = (idx < 4096) ? Wq : ((idx < 8192) ? Wk : Wv);
    int e = idx & 4095;
    sW[(idx >> 12) * 64 * LD64 + (e >> 6) * LD64 + (e & 63)] = (F16)src[e];
  }
  __syncthreads();

  // ---- phase 1: Qraw = Wq (64x64) x Pq^T (64x128) ------------------------
#pragma unroll
  for (int s = 0; s < 4; ++s) {
    const int tile = wave * 4 + s;            // 4 m-tiles x 8 n-tiles
    const int mt = tile >> 3, nt = tile & 7;
    v8f acc = {};
#pragma unroll
    for (int ks = 0; ks < 2; ++ks) {
      v16h a  = wmma_load_a (sW   + (mt * 16) * LD64 + ks * 32, LD64);
      v16h bb = wmma_load_bt(sPQt + (nt * 16) * LD64 + ks * 32, LD64);
      acc = WMMA_F32_F16(a, bb, acc);
    }
    const int n  = nt * 16 + (lane & 15);
    const int mb = mt * 16 + ((lane >> 4) ? 8 : 0);
#pragma unroll
    for (int v = 0; v < 8; ++v) sQF[(mb + v) * 128 + n] = acc[v];
  }
  __syncthreads();

  // ---- phase 2: Kraw = Wk x Pk^T (f32); V = Wv x Pk^T (f16, transposed) --
#pragma unroll
  for (int s = 0; s < 8; ++s) {
    const int tt   = wave * 8 + s;            // 0..31 K-proj, 32..63 V-proj
    const bool isV = tt >= 32;
    const int tile = tt & 31;
    const int mt = tile >> 3, nt = tile & 7;
    const F16* Wmat = sW + (isV ? 2 : 1) * 64 * LD64;
    v8f acc = {};
#pragma unroll
    for (int ks = 0; ks < 2; ++ks) {
      v16h a  = wmma_load_a (Wmat + (mt * 16) * LD64 + ks * 32, LD64);
      v16h bb = wmma_load_bt(sPKt + (nt * 16) * LD64 + ks * 32, LD64);
      acc = WMMA_F32_F16(a, bb, acc);
    }
    const int n  = nt * 16 + (lane & 15);
    const int mb = mt * 16 + ((lane >> 4) ? 8 : 0);
    if (isV) {
#pragma unroll
      for (int v = 0; v < 8; ++v) sVt[n * LD64 + (mb + v)] = (F16)acc[v];
    } else {
#pragma unroll
      for (int v = 0; v < 8; ++v) sKF[(mb + v) * 128 + n] = acc[v];
    }
  }
  __syncthreads();

  // ---- phase 3: row L2 norms (folded later) + f16 conversions ------------
  if (t < 64) {
    float ss = 0.f;
    const float* row = sQF + t * 128;
#pragma unroll 4
    for (int c = 0; c < 128; ++c) { float q = row[c]; ss += q * q; }
    sInvQ[t] = 1.0f / fmaxf(sqrtf(ss), 1e-12f);
  } else if (t < 128) {
    const int r = t - 64;
    float ss = 0.f;
    const float* row = sKF + r * 128;
#pragma unroll 4
    for (int c = 0; c < 128; ++c) { float k = row[c]; ss += k * k; }
    sInvK[r] = 1.0f / fmaxf(sqrtf(ss), 1e-12f);
  }
  for (int idx = t; idx < 8192; idx += 256) {
    int j = idx >> 7, c = idx & 127;
    sQh[j * LD128 + c] = (F16)sQF[idx];
    sKh[j * LD128 + c] = (F16)sKF[idx];   // natural [j][c] == Bt layout for QK^T
  }
  __syncthreads();

  // ---- phase 4: S = Qh (64x128) x Kh^T, scaled by norms*temp -------------
#pragma unroll
  for (int s = 0; s < 2; ++s) {
    const int tile = wave * 2 + s;            // 4 m-tiles x 4 n-tiles
    const int mt = tile >> 2, nt = tile & 3;
    v8f acc = {};
#pragma unroll
    for (int ks = 0; ks < 4; ++ks) {
      v16h a  = wmma_load_a (sQh + (mt * 16) * LD128 + ks * 32, LD128);
      v16h bb = wmma_load_bt(sKh + (nt * 16) * LD128 + ks * 32, LD128);
      acc = WMMA_F32_F16(a, bb, acc);
    }
    const int n  = nt * 16 + (lane & 15);
    const int mb = mt * 16 + ((lane >> 4) ? 8 : 0);
    const float sk = sInvK[n] * tempv;
#pragma unroll
    for (int v = 0; v < 8; ++v)
      sS[(mb + v) * 64 + n] = acc[v] * sInvQ[mb + v] * sk;
  }
  __syncthreads();

  // ---- phase 5: softmax over k (row per thread), write f16 attn ----------
  if (t < 64) {
    float* row = sS + t * 64;
    float mx = -3.4e38f;
#pragma unroll 4
    for (int k = 0; k < 64; ++k) mx = fmaxf(mx, row[k]);
    float sum = 0.f;
#pragma unroll 4
    for (int k = 0; k < 64; ++k) { float e = __expf(row[k] - mx); row[k] = e; sum += e; }
    const float inv = 1.0f / sum;
    F16* arow = sAttn + t * LD64;
#pragma unroll 4
    for (int k = 0; k < 64; ++k) arow[k] = (F16)(row[k] * inv);
  }
  __syncthreads();

  // ---- phase 6: O = attn (64x64) x V (64x128); scatter (unpatchify) ------
#pragma unroll
  for (int s = 0; s < 4; ++s) {
    const int tile = wave * 4 + s;            // 4 m-tiles x 8 n-tiles
    const int mt = tile >> 3, nt = tile & 7;
    v8f acc = {};
#pragma unroll
    for (int ks = 0; ks < 2; ++ks) {
      v16h a  = wmma_load_a (sAttn + (mt * 16) * LD64 + ks * 32, LD64);
      v16h bb = wmma_load_bt(sVt   + (nt * 16) * LD64 + ks * 32, LD64);
      acc = WMMA_F32_F16(a, bb, acc);
    }
    const int c  = nt * 16 + (lane & 15);
    const int mb = mt * 16 + ((lane >> 4) ? 8 : 0);
#pragma unroll
    for (int v = 0; v < 8; ++v) {
      const int j = mb + v;
      const int pi = j >> 3, pj = j & 7;
      long g = ((long)(b * 128 + c) * H + (pi * Hp + h)) * H + (pj * Hp + w);
      out[g] = acc[v];
    }
  }
}

// ---------------------------------------------------------------------------
// Precompute f16 A-matrix for the fused conv-transpose + concat + 1x1 conv:
//   Acomb[py][o][k]: k<128  -> Wproj[o][k]            (hi channels)
//                    k>=128 -> Weff[o][ci][py][px], k = 128 + ci*2 + px
//   beff[o] = sum_c Wproj[o][128+c] * bct[c]
// ---------------------------------------------------------------------------
__global__ void precompute_weights_kernel(const float* __restrict__ Wproj,
                                          const float* __restrict__ Wct,
                                          const float* __restrict__ bct,
                                          F16* __restrict__ Acomb,
                                          float* __restrict__ beff) {
  const int o = blockIdx.x;
  const int t = threadIdx.x;
  if (t < 128) {
    F16 wv = (F16)Wproj[o * 256 + t];
    Acomb[(0 * 128 + o) * 384 + t] = wv;
    Acomb[(1 * 128 + o) * 384 + t] = wv;
  }
  const int ci = t >> 1, px = t & 1;
#pragma unroll
  for (int py = 0; py < 2; ++py) {
    float sum = 0.f;
    for (int c = 0; c < 128; ++c)
      sum += Wproj[o * 256 + 128 + c] * Wct[((ci * 128 + c) * 2 + py) * 2 + px];
    Acomb[(py * 128 + o) * 384 + 128 + t] = (F16)sum;
  }
  if (o == 0 && t < 128) {
    float s = 0.f;
    for (int c = 0; c < 128; ++c) s += Wproj[t * 256 + 128 + c] * bct[c];
    beff[t] = s;
  }
}

// ---------------------------------------------------------------------------
// Fused conv-transpose(2,2) + concat + 1x1 conv as a 128 x 384 x 64 WMMA GEMM
// per (b, y, 64-wide x tile).  Lo channels are parity-expanded over x&1.
// B staged transposed [64n][392k] so each lane's B-fragment is two contiguous 16B runs.
// ---------------------------------------------------------------------------
#define LDF 392

__launch_bounds__(256)
__global__ void final_proj_kernel(const float* __restrict__ out_hi,
                                  const float* __restrict__ out_lo,
                                  const F16* __restrict__ Acomb,
                                  const float* __restrict__ beff,
                                  float* __restrict__ out) {
  __shared__ __align__(16) F16 sBt[64 * LDF];   // 49 KB
  const int blk = blockIdx.x;
  const int xt = blk & 3;
  const int y  = (blk >> 2) & 255;
  const int b  = blk >> 10;
  const int x0 = xt * 64;
  const int py = y & 1;
  const int t    = threadIdx.x;
  const int wave = t >> 5;
  const int lane = t & 31;

  for (int idx = t; idx < 128 * 64; idx += 256) {
    int k = idx >> 6, n = idx & 63;
    sBt[n * LDF + k] =
        (F16)out_hi[((long)(b * 128 + k) * 256 + y) * 256 + x0 + n];
  }
  for (int idx = t; idx < 256 * 64; idx += 256) {
    int kk = idx >> 6, n = idx & 63;
    int ci = kk >> 1, px = kk & 1;
    int x = x0 + n;
    float v = 0.f;
    if ((x & 1) == px)
      v = out_lo[((long)(b * 128 + ci) * 128 + (y >> 1)) * 128 + (x >> 1)];
    sBt[n * LDF + 128 + kk] = (F16)v;
  }
  __syncthreads();

  const F16* Ap = Acomb + (long)py * 128 * 384;   // L2-resident weights
#pragma unroll
  for (int s = 0; s < 4; ++s) {
    const int tile = wave * 4 + s;            // 8 m-tiles x 4 n-tiles
    const int mt = tile >> 2, nt = tile & 3;
    v8f acc = {};
#pragma unroll
    for (int ks = 0; ks < 12; ++ks) {
      v16h a  = wmma_load_a (Ap  + (mt * 16) * 384 + ks * 32, 384);
      v16h bb = wmma_load_bt(sBt + (nt * 16) * LDF + ks * 32, LDF);
      acc = WMMA_F32_F16(a, bb, acc);
    }
    const int x  = x0 + nt * 16 + (lane & 15);
    const int mb = mt * 16 + ((lane >> 4) ? 8 : 0);
#pragma unroll
    for (int v = 0; v < 8; ++v) {
      const int o = mb + v;
      out[((long)(b * 128 + o) * 256 + y) * 256 + x] = acc[v] + beff[o];
    }
  }
}

// ---------------------------------------------------------------------------
extern "C" void kernel_launch(void* const* d_in, const int* in_sizes, int n_in,
                              void* d_out, int out_size, void* d_ws, size_t ws_size,
                              hipStream_t stream) {
  const float* x_up   = (const float*)d_in[0];
  const float* x_down = (const float*)d_in[1];
  const float* Wq_hi  = (const float*)d_in[2];
  const float* Wk_hi  = (const float*)d_in[3];
  const float* Wv_hi  = (const float*)d_in[4];
  const float* Wq_lo  = (const float*)d_in[5];
  const float* Wk_lo  = (const float*)d_in[6];
  const float* Wv_lo  = (const float*)d_in[7];
  const float* temp   = (const float*)d_in[8];
  const float* Wct    = (const float*)d_in[9];
  const float* bct    = (const float*)d_in[10];
  const float* Wproj  = (const float*)d_in[11];
  float* out = (float*)d_out;

  char* ws = (char*)d_ws;
  float* out_hi    = (float*)(ws);                                   // 134,217,728 B
  float* x_down_lo = (float*)(ws + 134217728L);                      //  33,554,432 B
  float* x_up_lo   = (float*)(ws + 134217728L + 33554432L);          //  33,554,432 B
  float* out_lo    = (float*)(ws + 134217728L + 2L * 33554432L);     //  33,554,432 B
  F16*   Acomb     = (F16*)  (ws + 134217728L + 3L * 33554432L);     //     196,608 B
  float* beff      = (float*)(ws + 134217728L + 3L * 33554432L + 196608L);

  // 2x2 maxpool of both inputs -> lo-branch inputs
  maxpool2_kernel<<<32768, 256, 0, stream>>>(x_down, x_up, x_down_lo, x_up_lo,
                                             128, 128, 8388608L);
  // hi branch: q from x_down, k/v from x_up (H=256, Hp=32, 4096 patches)
  attn_branch_kernel<<<4096, 256, 0, stream>>>(x_down, x_up, Wq_hi, Wk_hi, Wv_hi,
                                               temp, out_hi, 256, 32);
  // lo branch (H=128, Hp=16, 1024 patches)
  attn_branch_kernel<<<1024, 256, 0, stream>>>(x_down_lo, x_up_lo, Wq_lo, Wk_lo, Wv_lo,
                                               temp, out_lo, 128, 16);
  // fold ConvTranspose + concat + 1x1 conv weights
  precompute_weights_kernel<<<128, 256, 0, stream>>>(Wproj, Wct, bct, Acomb, beff);
  // fused epilogue GEMM
  final_proj_kernel<<<4096, 256, 0, stream>>>(out_hi, out_lo, Acomb, beff, out);
}
